// DocSenModel_14001593385384
// MI455X (gfx1250) — matrix operand order repaired and checked by reading, not compiled, not run on hardware
//
#include <hip/hip_runtime.h>
#include <hip/hip_bf16.h>

// ---- CDNA5 WMMA types ----
typedef __attribute__((ext_vector_type(16))) _Float16 v16h;
typedef __attribute__((ext_vector_type(8)))  _Float16 v8h;
typedef __attribute__((ext_vector_type(8)))  float    v8f;

// Problem constants (match reference)
#define VOCAB 100000
#define DD    200     // embedding dim
#define HH    50      // hidden
#define SS    128     // sentences
#define WW    64      // words per sentence
#define NOUT  5

// Padded GEMM dims
#define MP    64      // H padded to 4x16
#define KP    224     // D padded to 7x32
#define BSTR  232     // LDS B row stride (halfwords), 464B = 16B aligned, conflict-free
#define CSTR  66      // LDS C row stride (floats)

// Workspace layout (bytes)
#define WS_A_OFF    0                      // f16 A [64][224]  = 28672 B
#define WS_REPS_OFF 32768                  // f32 reps [128][50] = 25600 B

// ------------------------------------------------------------------
// K0: pack w_word [50][200] f32 -> padded f16 A [64][224] in ws
// ------------------------------------------------------------------
__global__ void k0_pack_a(const float* __restrict__ w_word,
                          _Float16* __restrict__ Af16) {
    int t = blockIdx.x * blockDim.x + threadIdx.x;
    if (t >= MP * KP) return;
    int m = t / KP, k = t % KP;
    float v = (m < HH && k < DD) ? w_word[m * DD + k] : 0.f;
    Af16[t] = (_Float16)v;
}

// ------------------------------------------------------------------
// K1: per-sentence fused kernel (1 block = 1 sentence, 256 thr = 8 waves)
//   gather emb rows -> f16 B in LDS
//   WMMA: C[64][64] = A[64x224] * B[224x64]   (x[s,h,w] before bias)
//   reduce columns -> A1, x0, x1, x62, x63 per channel (bias folded in)
//   conv-branch tanh means -> reps[s][50]
// ------------------------------------------------------------------
__global__ void __launch_bounds__(256)
k1_sentence(const int*   __restrict__ doc,
            const float* __restrict__ emb,
            const float* __restrict__ b_word,
            const float* __restrict__ w1, const float* __restrict__ b1,
            const float* __restrict__ w2, const float* __restrict__ b2,
            const float* __restrict__ w3, const float* __restrict__ b3,
            const _Float16* __restrict__ Af16,
            float* __restrict__ reps) {
    __shared__ _Float16 Bt[WW * BSTR];      // [word][k] f16, 29696 B
    __shared__ float    Cls[MP * CSTR];     // [h][w] f32, 16896 B
    __shared__ float    part[MP][4];
    __shared__ float    A1s[MP], X0s[MP], X1s[MP], X62s[MP], X63s[MP];

    const int s   = blockIdx.x;
    const int tid = threadIdx.x;

    // ---- stage gathered embeddings as f16 into LDS (B matrix, [w][k]) ----
    {
        int w = tid >> 2, q = tid & 3;               // 4 threads per word
        int idx = doc[s * WW + w];
        const float* er = emb + (size_t)idx * DD;
        #pragma unroll 1
        for (int d = q; d < KP; d += 4) {
            float v = (d < DD) ? er[d] : 0.f;
            Bt[w * BSTR + d] = (_Float16)v;
        }
    }
    __syncthreads();

    // ---- WMMA: each wave owns mtile = wave&3, ntiles {2*(wave>>2), +1} ----
    {
        const int wave = tid >> 5;
        const int lane = tid & 31;
        const int half = lane >> 4;          // 0: lanes 0-15, 1: lanes 16-31
        const int l16  = lane & 15;
        const int mt   = wave & 3;
        const int nt0  = (wave >> 2) * 2;
        const int nt1  = nt0 + 1;
        const int mrow = mt * 16 + l16;

        v8f c0 = {}, c1 = {};
        #pragma unroll
        for (int kt = 0; kt < 7; ++kt) {
            // A fragment: elem j -> K = kt*32 + half*16 + j
            const _Float16* ab = Af16 + mrow * KP + kt * 32 + half * 16;
            v8h alo = *reinterpret_cast<const v8h*>(ab);
            v8h ahi = *reinterpret_cast<const v8h*>(ab + 8);
            v16h a;
            #pragma unroll
            for (int j = 0; j < 8; ++j) { a[j] = alo[j]; a[j + 8] = ahi[j]; }

            // B fragments: elem j -> K = kt*32 + half*16 + j, col = nt*16+l16
            const int kb = kt * 32 + half * 16;
            const _Float16* bb0 = &Bt[(nt0 * 16 + l16) * BSTR + kb];
            const _Float16* bb1 = &Bt[(nt1 * 16 + l16) * BSTR + kb];
            v8h b0lo = *reinterpret_cast<const v8h*>(bb0);
            v8h b0hi = *reinterpret_cast<const v8h*>(bb0 + 8);
            v8h b1lo = *reinterpret_cast<const v8h*>(bb1);
            v8h b1hi = *reinterpret_cast<const v8h*>(bb1 + 8);
            v16h b0, b1;
            #pragma unroll
            for (int j = 0; j < 8; ++j) {
                b0[j] = b0lo[j]; b0[j + 8] = b0hi[j];
                b1[j] = b1lo[j]; b1[j + 8] = b1hi[j];
            }

            c0 = __builtin_amdgcn_wmma_f32_16x16x32_f16(
                     false, a, false, b0, (short)0, c0, false, false);
            c1 = __builtin_amdgcn_wmma_f32_16x16x32_f16(
                     false, a, false, b1, (short)0, c1, false, false);
        }

        // ---- C tiles -> LDS: VGPR r holds (M = r + 8*half, N = l16) ----
        #pragma unroll
        for (int r = 0; r < 8; ++r) {
            int row = mt * 16 + half * 8 + r;
            Cls[row * CSTR + nt0 * 16 + l16] = c0[r];
            Cls[row * CSTR + nt1 * 16 + l16] = c1[r];
        }
    }
    __syncthreads();

    // ---- column stats: rowsum over w, and endpoints (fold in b_word) ----
    {
        int row = tid >> 2, q = tid & 3;
        float acc = 0.f;
        #pragma unroll
        for (int n = 0; n < 16; ++n) acc += Cls[row * CSTR + q * 16 + n];
        part[row][q] = acc;
    }
    __syncthreads();
    if (tid < MP) {
        int m = tid;
        float bw = (m < HH) ? b_word[m] : 0.f;
        float rs = part[m][0] + part[m][1] + part[m][2] + part[m][3];
        A1s[m]  = rs + 64.f * bw;
        X0s[m]  = Cls[m * CSTR + 0]  + bw;
        X1s[m]  = Cls[m * CSTR + 1]  + bw;
        X62s[m] = Cls[m * CSTR + 62] + bw;
        X63s[m] = Cls[m * CSTR + 63] + bw;
    }
    __syncthreads();

    // ---- conv branches (mean over time == window sums) + tanh + average ----
    if (tid < HH) {
        int h = tid;
        float a1 = 0.f, a2 = 0.f, a3 = 0.f;
        #pragma unroll 1
        for (int i = 0; i < HH; ++i) {
            float A1 = A1s[i], x0 = X0s[i], x1 = X1s[i],
                  x62 = X62s[i], x63 = X63s[i];
            a1 += w1[h * HH + i] * A1;
            // K=2: windows [0..62]=A1-x63, [1..63]=A1-x0
            a2 += w2[(h * HH + i) * 2 + 0] * (A1 - x63)
                + w2[(h * HH + i) * 2 + 1] * (A1 - x0);
            // K=3: [0..61]=A1-x62-x63, [1..62]=A1-x0-x63, [2..63]=A1-x0-x1
            a3 += w3[(h * HH + i) * 3 + 0] * (A1 - x62 - x63)
                + w3[(h * HH + i) * 3 + 1] * (A1 - x0 - x63)
                + w3[(h * HH + i) * 3 + 2] * (A1 - x0 - x1);
        }
        float r1 = tanhf(a1 / 64.f + b1[h]);
        float r2 = tanhf(a2 / 63.f + b2[h]);
        float r3 = tanhf(a3 / 62.f + b3[h]);
        reps[s * HH + h] = (r1 + r2 + r3) * (1.f / 3.f);
    }
}

// ------------------------------------------------------------------
// K2: sequential GRNN scan (1 block, 64 threads) + softmax head
// ------------------------------------------------------------------
__global__ void __launch_bounds__(64)
k2_scan(const float* __restrict__ reps,
        const float* __restrict__ Wi, const float* __restrict__ Ui, const float* __restrict__ bi,
        const float* __restrict__ Wf, const float* __restrict__ Uf, const float* __restrict__ bf,
        const float* __restrict__ Wr, const float* __restrict__ Ur, const float* __restrict__ br,
        const float* __restrict__ Wo, const float* __restrict__ bo,
        float* __restrict__ out) {
    __shared__ float hprev[HH];
    const int h = threadIdx.x;
    if (h < HH) hprev[h] = 0.f;

    for (int s = 0; s < SS; ++s) {
        __syncthreads();
        float hnew = 0.f;
        if (h < HH) {
            float si = bi[h], sf = bf[h], sr = br[h];
            #pragma unroll 1
            for (int j = 0; j < HH; ++j) {
                float r  = reps[s * HH + j];
                float hp = hprev[j];
                si += Wi[h * HH + j] * r + Ui[h * HH + j] * hp;
                sf += Wf[h * HH + j] * r + Uf[h * HH + j] * hp;
                sr += Wr[h * HH + j] * r + Ur[h * HH + j] * hp;
            }
            float ig = 1.f / (1.f + __expf(-si));
            float fg = 1.f / (1.f + __expf(-sf));
            float gg = tanhf(sr);
            hnew = tanhf(ig * gg + fg * hprev[h]);
        }
        __syncthreads();
        if (h < HH) hprev[h] = hnew;
    }
    __syncthreads();

    if (h == 0) {
        float lg[NOUT];
        float mx = -1e30f;
        #pragma unroll
        for (int o = 0; o < NOUT; ++o) {
            float acc = bo[o];
            for (int j = 0; j < HH; ++j) acc += Wo[o * HH + j] * hprev[j];
            lg[o] = acc;
            mx = fmaxf(mx, acc);
        }
        float den = 0.f;
        #pragma unroll
        for (int o = 0; o < NOUT; ++o) { lg[o] = __expf(lg[o] - mx); den += lg[o]; }
        #pragma unroll
        for (int o = 0; o < NOUT; ++o) out[o] = lg[o] / den;
    }
}

// ------------------------------------------------------------------
extern "C" void kernel_launch(void* const* d_in, const int* in_sizes, int n_in,
                              void* d_out, int out_size, void* d_ws, size_t ws_size,
                              hipStream_t stream) {
    const int*   doc    = (const int*)  d_in[0];
    const float* emb    = (const float*)d_in[1];
    const float* w_word = (const float*)d_in[2];
    const float* b_word = (const float*)d_in[3];
    const float* cw1    = (const float*)d_in[4];
    const float* cb1    = (const float*)d_in[5];
    const float* cw2    = (const float*)d_in[6];
    const float* cb2    = (const float*)d_in[7];
    const float* cw3    = (const float*)d_in[8];
    const float* cb3    = (const float*)d_in[9];
    const float* Wi     = (const float*)d_in[10];
    const float* Ui     = (const float*)d_in[11];
    const float* bi     = (const float*)d_in[12];
    const float* Wf     = (const float*)d_in[13];
    const float* Uf     = (const float*)d_in[14];
    const float* bf     = (const float*)d_in[15];
    const float* Wr     = (const float*)d_in[16];
    const float* Ur     = (const float*)d_in[17];
    const float* br     = (const float*)d_in[18];
    const float* Wo     = (const float*)d_in[19];
    const float* bo     = (const float*)d_in[20];

    _Float16* Af16 = (_Float16*)((char*)d_ws + WS_A_OFF);
    float*    reps = (float*)   ((char*)d_ws + WS_REPS_OFF);
    float*    out  = (float*)d_out;

    k0_pack_a<<<(MP * KP + 255) / 256, 256, 0, stream>>>(w_word, Af16);
    k1_sentence<<<SS, 256, 0, stream>>>(doc, emb, b_word,
                                        cw1, cb1, cw2, cb2, cw3, cb3,
                                        Af16, reps);
    k2_scan<<<1, 64, 0, stream>>>(reps, Wi, Ui, bi, Wf, Uf, bf, Wr, Ur, br,
                                  Wo, bo, out);
}
